// ContextualAttentionModule_60559038874149
// MI455X (gfx1250) — matrix operand order repaired and checked by reading, not compiled, not run on hardware
//
#include <hip/hip_runtime.h>
#include <hip/hip_bf16.h>
#include <math.h>

// ---------------------------------------------------------------------------
// ContextualAttention (DeepFill) for MI455X / gfx1250, wave32 + WMMA f16->f32.
// All three heavy GEMMs go through v_wmma_f32_16x16x32_f16.
// ---------------------------------------------------------------------------

typedef __attribute__((ext_vector_type(16))) _Float16 v16h;
typedef __attribute__((ext_vector_type(8)))  _Float16 v8h;
typedef __attribute__((ext_vector_type(8)))  float    v8f;

#define CCH    128      // channels
#define IMG    128      // H = W
#define DS     64       // downsampled H = W
#define NP     4096     // 64*64 patches / queries
#define K1     1152     // 128 * 3 * 3
#define CK     2048     // 128 * 4 * 4
#define QF     16384    // 128 * 128
#define SCALE_F 10.0f

// -------------------------- WMMA fragment loads ----------------------------
// A-matrix 16x32 f16 layout (ISA 7.12.2): lane<16 holds M=lane, K = k0..k0+7
// and k0+16..k0+23 with k0 = 0; lane>=16 same rows of K base 8. Two b128 loads.
__device__ __forceinline__ v16h load_frag_a(const _Float16* __restrict__ base,
                                            int ldk, int row0, int kk, int lane) {
  const int r = lane & 15, h = lane >> 4;
  const _Float16* p = base + (size_t)(row0 + r) * ldk + kk + h * 8;
  v8h lo = *(const v8h*)(p);
  v8h hi = *(const v8h*)(p + 16);
  v16h f;
#pragma unroll
  for (int e = 0; e < 8; ++e) { f[e] = lo[e]; f[e + 8] = hi[e]; }
  return f;
}

// B-matrix 32x16 f16: lane<16 holds N=lane with K 0..7 and 16..23; lane>=16
// holds K 8..15 and 24..31 (same K order as A). Two b128 loads.
__device__ __forceinline__ v16h load_frag_b(const _Float16* __restrict__ base,
                                            int ldk, int row0, int kk, int lane) {
  const int r = lane & 15, h = lane >> 4;
  const _Float16* p = base + (size_t)(row0 + r) * ldk + kk + h * 8;
  v8h lo = *(const v8h*)(p);
  v8h hi = *(const v8h*)(p + 16);
  v16h f;
#pragma unroll
  for (int e = 0; e < 8; ++e) { f[e] = lo[e]; f[e + 8] = hi[e]; }
  return f;
}

// Generic GEMM: C[M x N] = A[M x K] * B[N x K]^T, f16 in, f32 accumulate.
// One wave per 64x64 C tile (4x4 WMMA accumulators). Operands are L2-resident
// (<=36 MiB each vs 192 MB L2) so fragments load straight from global.
// mode 0: C[m*N+n] = acc * rowscale[m]            (rowscale may be null)
// mode 1: C[m*N+n] = elu(acc + bias[n])
// mode 2: C[n*M+m] = elu(acc + bias[n])           (NCHW-transposed store)
__global__ __launch_bounds__(256)
void wmma_gemm_kernel(const _Float16* __restrict__ A, const _Float16* __restrict__ B,
                      float* __restrict__ C, int M, int N, int K,
                      const float* __restrict__ rowscale,
                      const float* __restrict__ bias, int mode) {
  const int lane   = threadIdx.x & 31;
  const int wave   = blockIdx.x * (blockDim.x >> 5) + (threadIdx.x >> 5);
  const int tilesN = N >> 6;
  const int tiles  = (M >> 6) * tilesN;
  if (wave >= tiles) return;
  const int tm = wave / tilesN, tn = wave - tm * tilesN;
  const int m0 = tm << 6, n0 = tn << 6;

  v8f zero;
#pragma unroll
  for (int e = 0; e < 8; ++e) zero[e] = 0.0f;
  v8f acc[4][4];
#pragma unroll
  for (int i = 0; i < 4; ++i)
#pragma unroll
    for (int j = 0; j < 4; ++j) acc[i][j] = zero;

  for (int kk = 0; kk < K; kk += 32) {
    v16h af[4], bf[4];
#pragma unroll
    for (int i = 0; i < 4; ++i) af[i] = load_frag_a(A, K, m0 + i * 16, kk, lane);
#pragma unroll
    for (int j = 0; j < 4; ++j) bf[j] = load_frag_b(B, K, n0 + j * 16, kk, lane);
#pragma unroll
    for (int i = 0; i < 4; ++i)
#pragma unroll
      for (int j = 0; j < 4; ++j)
        acc[i][j] = __builtin_amdgcn_wmma_f32_16x16x32_f16(
            false, af[i], false, bf[j], (short)0, acc[i][j], false, false);
  }

  // C/D layout: value v of a tile lives at m = base_m + v + 8*(lane>=16),
  // n = base_n + (lane & 15).
  const int cn = lane & 15;
  const int mh = (lane >> 4) * 8;
#pragma unroll
  for (int i = 0; i < 4; ++i) {
#pragma unroll
    for (int j = 0; j < 4; ++j) {
      const int n = n0 + j * 16 + cn;
#pragma unroll
      for (int v = 0; v < 8; ++v) {
        const int m = m0 + i * 16 + mh + v;
        float x = acc[i][j][v];
        if (mode == 0) {
          if (rowscale) x *= rowscale[m];
          C[(size_t)m * N + n] = x;
        } else if (mode == 1) {
          x += bias[n];
          x = x > 0.0f ? x : (expf(x) - 1.0f);
          C[(size_t)m * N + n] = x;
        } else {
          x += bias[n];
          x = x > 0.0f ? x : (expf(x) - 1.0f);
          C[(size_t)n * M + m] = x;
        }
      }
    }
  }
}

// -------- im2col of a stride-2 downsampled image, 3x3 window, pad 1 --------
// dst[q=(py*64+px)][k=(c*9+dy*3+dx)] = src_ds[c, py+dy-1, px+dx-1] (f16)
__global__ void im2col_ds_kernel(const float* __restrict__ src,
                                 _Float16* __restrict__ dst, int batch) {
  int idx = blockIdx.x * blockDim.x + threadIdx.x;
  if (idx >= NP * K1) return;
  int q = idx / K1, k = idx - q * K1;
  int c = k / 9, r9 = k - c * 9;
  int dy = r9 / 3, dx = r9 - dy * 3;
  int py = q >> 6, px = q & 63;
  int yy = py + dy - 1, xx = px + dx - 1;
  float v = 0.0f;
  if ((unsigned)yy < DS && (unsigned)xx < DS)
    v = src[(((size_t)batch * CCH + c) * IMG + 2 * yy) * IMG + 2 * xx];
  dst[idx] = (_Float16)v;
}

// ----- per-patch 1/max(||patch||,1e-4) in f32 (one wave per patch) ---------
__global__ void patch_norm_kernel(const float* __restrict__ bg,
                                  float* __restrict__ invn, int batch) {
  int lane = threadIdx.x & 31;
  int n = blockIdx.x * (blockDim.x >> 5) + (threadIdx.x >> 5);
  if (n >= NP) return;
  int py = n >> 6, px = n & 63;
  float s = 0.0f;
  for (int t = lane; t < K1; t += 32) {
    int c = t / 9, r9 = t - c * 9;
    int dy = r9 / 3, dx = r9 - dy * 3;
    int yy = py + dy - 1, xx = px + dx - 1;
    if ((unsigned)yy < DS && (unsigned)xx < DS) {
      float v = bg[(((size_t)batch * CCH + c) * IMG + 2 * yy) * IMG + 2 * xx];
      s += v * v;
    }
  }
#pragma unroll
  for (int off = 16; off > 0; off >>= 1) s += __shfl_xor(s, off, 32);
  if (lane == 0) invn[n] = 1.0f / fmaxf(sqrtf(s), 1e-4f);
}

// ------------- eq_mask[n] = (mean of 3x3 window of mask_ds == 0) -----------
__global__ void emask_kernel(const float* __restrict__ mask, float* __restrict__ em) {
  int n = blockIdx.x * blockDim.x + threadIdx.x;
  if (n >= NP) return;
  int py = n >> 6, px = n & 63;
  float s = 0.0f;
  for (int dy = 0; dy < 3; ++dy)
    for (int dx = 0; dx < 3; ++dx) {
      int yy = py + dy - 1, xx = px + dx - 1;
      if ((unsigned)yy < DS && (unsigned)xx < DS)
        s += mask[(2 * yy) * IMG + 2 * xx];
    }
  em[n] = (s == 0.0f) ? 1.0f : 0.0f;
}

// ---- fuse pass 1: identity-diagonal 3x3 over flat (4096 x 4096) plane -----
__global__ void fuse1_kernel(const float* __restrict__ S, float* __restrict__ D) {
  size_t idx = (size_t)blockIdx.x * blockDim.x + threadIdx.x;
  int r = (int)(idx >> 12), c = (int)(idx & 4095);
  float v = S[idx];
  if (r > 0 && c > 0) v += S[idx - 4097];
  if (r < 4095 && c < 4095) v += S[idx + 4097];
  D[idx] = v;
}

// ---- fuse pass 2: same diag conv in the (bx,by)/(fx,fy)-transposed layout.
// Output is stored directly in that transposed layout [r2=(bx*64+by)][c2=(fx*64+fy)]
// which is exactly what softmax + GEMM2 consume.
__device__ __forceinline__ int tp64(int i) { return ((i & 63) << 6) | (i >> 6); }
__global__ void fuse2_kernel(const float* __restrict__ S, float* __restrict__ D) {
  size_t idx = (size_t)blockIdx.x * blockDim.x + threadIdx.x;
  int r2 = (int)(idx >> 12), c2 = (int)(idx & 4095);
  float v = 0.0f;
#pragma unroll
  for (int d = -1; d <= 1; ++d) {
    int rr = r2 + d, cc = c2 + d;
    if ((unsigned)rr < NP && (unsigned)cc < NP)
      v += S[(size_t)tp64(rr) * NP + tp64(cc)];
  }
  D[idx] = v;
}

// ---- softmax over the 4096 patch channels; writes f16 A[c2][r2] (K-major) --
__global__ __launch_bounds__(256)
void softmax_kernel(const float* __restrict__ S, const float* __restrict__ em,
                    _Float16* __restrict__ A) {
  __shared__ float red[4][64];
  const int cid = threadIdx.x & 63;
  const int rg  = threadIdx.x >> 6;
  const int col = blockIdx.x * 64 + cid;

  float mx = -3.4e38f;
  for (int r = rg; r < NP; r += 4) {
    float t = S[(size_t)r * NP + col] * em[r] * SCALE_F;
    mx = fmaxf(mx, t);
  }
  red[rg][cid] = mx;
  __syncthreads();
  mx = fmaxf(fmaxf(red[0][cid], red[1][cid]), fmaxf(red[2][cid], red[3][cid]));
  __syncthreads();

  float sm = 0.0f;
  for (int r = rg; r < NP; r += 4) {
    float t = S[(size_t)r * NP + col] * em[r] * SCALE_F;
    sm += expf(t - mx);
  }
  red[rg][cid] = sm;
  __syncthreads();
  sm = red[0][cid] + red[1][cid] + red[2][cid] + red[3][cid];
  float inv = 1.0f / sm;

  for (int r = rg; r < NP; r += 4) {
    float t = S[(size_t)r * NP + col] * em[r] * SCALE_F;
    float p = expf(t - mx) * inv * em[r];
    A[(size_t)col * NP + r] = (_Float16)p;
  }
}

// ---- raw 4x4 stride-2 patches as B-operand: R[ck=(c*16+ky*4+kx)][r2] ------
__global__ void prep_raw_kernel(const float* __restrict__ bg,
                                _Float16* __restrict__ R, int batch) {
  int idx = blockIdx.x * blockDim.x + threadIdx.x;
  if (idx >= CK * NP) return;
  int ck = idx >> 12, r2 = idx & 4095;
  int c = ck >> 4, kk = ck & 15, ky = kk >> 2, kx = kk & 3;
  int py = r2 >> 6, px = r2 & 63;
  int yy = 2 * py + ky - 1, xx = 2 * px + kx - 1;
  float v = 0.0f;
  if ((unsigned)yy < IMG && (unsigned)xx < IMG)
    v = bg[(((size_t)batch * CCH + c) * IMG + yy) * IMG + xx];
  R[idx] = (_Float16)v;
}

// ---- overlap-add fold of T[q'=(fx*64+fy)][c*16+ky*4+kx] into y (gather) ----
// Transposed-conv relation: oy = 2*fx + ky - 1, ox = 2*fy + kx - 1.
__global__ void fold_kernel(const float* __restrict__ T, float* __restrict__ Y,
                            int batch) {
  int idx = blockIdx.x * blockDim.x + threadIdx.x;
  if (idx >= CCH * QF) return;
  int c = idx / QF, rem = idx - c * QF;
  int oy = rem >> 7, ox = rem & 127;
  float acc = 0.0f;
  int ky0 = (oy + 1) & 1, kx0 = (ox + 1) & 1;
#pragma unroll
  for (int a = 0; a < 2; ++a) {
    int ky = ky0 + 2 * a;
    int fx = (oy + 1 - ky) >> 1;
    if ((unsigned)fx >= DS) continue;
#pragma unroll
    for (int bq = 0; bq < 2; ++bq) {
      int kx = kx0 + 2 * bq;
      int fy = (ox + 1 - kx) >> 1;
      if ((unsigned)fy >= DS) continue;
      acc += T[(size_t)(fx * 64 + fy) * CK + (c << 4) + (ky << 2) + kx];
    }
  }
  Y[((size_t)batch * CCH + c) * QF + rem] = 0.25f * acc;
}

// ---- im2col (3x3, pad 1) for the final convs; layout 0 = CHW, 1 = HWC -----
__global__ void im2col_img_kernel(const float* __restrict__ src,
                                  _Float16* __restrict__ dst, int layout) {
  int idx = blockIdx.x * blockDim.x + threadIdx.x;
  if (idx >= QF * K1) return;
  int q = idx / K1, k = idx - q * K1;
  int c = k / 9, r9 = k - c * 9;
  int dy = r9 / 3, dx = r9 - dy * 3;
  int y = q >> 7, x = q & 127;
  int yy = y + dy - 1, xx = x + dx - 1;
  float v = 0.0f;
  if ((unsigned)yy < IMG && (unsigned)xx < IMG)
    v = (layout == 0) ? src[((size_t)c * IMG + yy) * IMG + xx]
                      : src[(size_t)(yy * IMG + xx) * CCH + c];
  dst[idx] = (_Float16)v;
}

__global__ void f32_to_f16_kernel(const float* __restrict__ src,
                                  _Float16* __restrict__ dst, int n) {
  int i = blockIdx.x * blockDim.x + threadIdx.x;
  if (i < n) dst[i] = (_Float16)src[i];
}

// ---------------------------------------------------------------------------
extern "C" void kernel_launch(void* const* d_in, const int* in_sizes, int n_in,
                              void* d_out, int out_size, void* d_ws, size_t ws_size,
                              hipStream_t stream) {
  const float* fg   = (const float*)d_in[0];
  const float* bg   = (const float*)d_in[1];
  const float* mask = (const float*)d_in[2];
  const float* w1   = (const float*)d_in[3];
  const float* b1   = (const float*)d_in[4];
  const float* w2   = (const float*)d_in[5];
  const float* b2   = (const float*)d_in[6];
  float* out = (float*)d_out;
  char*  ws  = (char*)d_ws;

  // Workspace layout (bytes); ~211 MiB peak, regions reused across stages.
  _Float16* Q   = (_Float16*)(ws + 0);           //  9.44 MB  query im2col
  _Float16* Km  = (_Float16*)(ws + 9437184);     //  9.44 MB  key im2col
  float*    inv = (float*)   (ws + 18874368);    //  16 KB    1/patch-norm
  float*    em  = (float*)   (ws + 18890752);    //  16 KB    eq_mask
  _Float16* W1h = (_Float16*)(ws + 18907136);    //  288 KB
  _Float16* W2h = (_Float16*)(ws + 19202048);    //  288 KB
  float*    S0  = (float*)   (ws + 19496960);    //  64 MB    scores / S2 / X
  float*    S1  = (float*)   (ws + 86605824);    //  64 MB    fused / T / H
  _Float16* Am  = (_Float16*)(ws + 153714688);   //  32 MB    softmax weights
  _Float16* Rm  = (_Float16*)(ws + 187269120);   //  16 MB    raw patches
  float*    Y   = (float*)   (ws + 204046336);   //  16 MB    attended (2 batches)
  _Float16* X   = (_Float16*)S0;                 // alias: final-conv im2col
  float*    T   = S1;                            // alias: paste-GEMM output
  float*    Hb  = S1;                            // alias: hidden activations

  const int TB = 256;

  emask_kernel<<<(NP + TB - 1) / TB, TB, 0, stream>>>(mask, em);
  f32_to_f16_kernel<<<(CCH * K1 + TB - 1) / TB, TB, 0, stream>>>(w1, W1h, CCH * K1);
  f32_to_f16_kernel<<<(CCH * K1 + TB - 1) / TB, TB, 0, stream>>>(w2, W2h, CCH * K1);

  for (int b = 0; b < 2; ++b) {
    // Stage 1: im2col + norms + score GEMM  S0[n][q] = invn[n] * K.Q
    im2col_ds_kernel<<<(NP * K1 + TB - 1) / TB, TB, 0, stream>>>(fg, Q, b);
    im2col_ds_kernel<<<(NP * K1 + TB - 1) / TB, TB, 0, stream>>>(bg, Km, b);
    patch_norm_kernel<<<NP / 8, TB, 0, stream>>>(bg, inv, b);
    {
      int tiles = (NP / 64) * (NP / 64);   // 4096 waves
      wmma_gemm_kernel<<<(tiles + 7) / 8, TB, 0, stream>>>(
          Km, Q, S0, NP, NP, K1, inv, nullptr, 0);
    }

    // Stage 2: double diagonal fuse (with layout transpose) + softmax -> f16 A
    fuse1_kernel<<<65536, TB, 0, stream>>>(S0, S1);
    fuse2_kernel<<<65536, TB, 0, stream>>>(S1, S0);   // S2 -> S0 (transposed layout)
    softmax_kernel<<<NP / 64, TB, 0, stream>>>(S0, em, Am);

    // Stage 3: paste GEMM  T[q'][ck] = A.R  then overlap-add fold into Y
    prep_raw_kernel<<<(CK * NP + TB - 1) / TB, TB, 0, stream>>>(bg, Rm, b);
    {
      int tiles = (NP / 64) * (CK / 64);   // 2048 waves
      wmma_gemm_kernel<<<(tiles + 7) / 8, TB, 0, stream>>>(
          Am, Rm, T, NP, CK, NP, nullptr, nullptr, 0);
    }
    fold_kernel<<<(CCH * QF + TB - 1) / TB, TB, 0, stream>>>(T, Y, b);
  }

  // Stage 4: two 3x3 conv + ELU layers as WMMA GEMMs
  for (int b = 0; b < 2; ++b) {
    const float* yb = Y + (size_t)b * CCH * QF;
    im2col_img_kernel<<<(QF * K1 + TB - 1) / TB, TB, 0, stream>>>(yb, X, 0);
    {
      int tiles = (QF / 64) * (CCH / 64);  // 512 waves
      wmma_gemm_kernel<<<(tiles + 7) / 8, TB, 0, stream>>>(
          X, W1h, Hb, QF, CCH, K1, nullptr, b1, 1);
    }
    im2col_img_kernel<<<(QF * K1 + TB - 1) / TB, TB, 0, stream>>>(Hb, X, 1);
    {
      int tiles = (QF / 64) * (CCH / 64);
      wmma_gemm_kernel<<<(tiles + 7) / 8, TB, 0, stream>>>(
          X, W2h, out + (size_t)b * CCH * QF, QF, CCH, K1, nullptr, b2, 2);
    }
  }
}
